// AttackHead_84189948936884
// MI455X (gfx1250) — matrix operand order repaired and checked by reading, not compiled, not run on hardware
//
#include <hip/hip_runtime.h>

typedef __attribute__((ext_vector_type(16))) __bf16 v16bf;
typedef __attribute__((ext_vector_type(8)))  float  v8f;

#define NE        500000
#define MTILES    31250                         // NE / 16
#define MT        4                             // M-tiles (of 16 edges) per wave
#define WAVES     4                             // waves per workgroup
#define NJOBS     ((MTILES + MT - 1) / MT)      // 7813
#define GRID_MAIN ((NJOBS + WAVES - 1) / WAVES) // 1954
#define LDS_A     (MT * 16 * 1024)              // 64 KB per wave: 64 rows x 512 bf16
#define LDS_HA    4096                          // 16 x 128 bf16 scratch per wave
#define LDS_TOTAL (WAVES * LDS_A + WAVES * LDS_HA) // 278528 B < 320 KB

__device__ __forceinline__ unsigned f2bf(float f) {
  unsigned u = __float_as_uint(f);
  return (u + 0x7FFFu + ((u >> 16) & 1u)) >> 16;   // round-to-nearest-even
}

// ---- node_embeddings f32 -> bf16 table (row-major, 256 bf16 per node) ----
__global__ void conv_emb_kernel(const float4* __restrict__ src,
                                uint2* __restrict__ dst, int n4) {
  int t = blockIdx.x * blockDim.x + threadIdx.x;
  if (t >= n4) return;
  float4 v = src[t];
  dst[t] = make_uint2(f2bf(v.x) | (f2bf(v.y) << 16),
                      f2bf(v.z) | (f2bf(v.w) << 16));
}

// ---- pack row-major f32 weight [K][N] into per-lane WMMA B fragments ----
// B tile = 32x16 (KxN) bf16: lanes 0-15 -> N=r, K=0..15; lanes 16-31 -> N=r, K=16..31
// per lane 8 dwords, dword j = {K=2j (lo), K=2j+1 (hi)} within the lane's K window.
// Tile storage: 1 KB contiguous, lane L at byte L*32, tiles indexed kt*NT + nt.
__global__ void pack_w_kernel(const float* __restrict__ src, int K, int N,
                              unsigned* __restrict__ dst) {
  int t = blockIdx.x * blockDim.x + threadIdx.x;
  int NT = N >> 4;
  int total = (K >> 5) * NT * 256;
  if (t >= total) return;
  int tile = t >> 8;
  int w = t & 255;
  int j = w & 7;
  int L = w >> 3;
  int r = L & 15, hi = L >> 4;
  int kt = tile / NT, nt = tile - kt * NT;
  int k0 = kt * 32 + hi * 16 + 2 * j;
  int n  = nt * 16 + r;
  unsigned lo = f2bf(src[(size_t)k0 * N + n]);
  unsigned hh = f2bf(src[(size_t)(k0 + 1) * N + n]);
  dst[(size_t)tile * 256 + L * 8 + j] = lo | (hh << 16);
}

union ABu { uint4 q[2]; v16bf v; };

__global__ __launch_bounds__(WAVES * 32, 1)
void attack_head_main(const uint4* __restrict__ embq,   // bf16 table, 32 uint4 per node
                      const int*   __restrict__ edges,  // int64 pairs viewed as int32
                      const int*   __restrict__ army,
                      const uint4* __restrict__ w1p,
                      const uint4* __restrict__ wa1p,
                      const uint4* __restrict__ wa2p,
                      const float* __restrict__ b1,
                      const float* __restrict__ W2,
                      const float* __restrict__ b2,
                      const float* __restrict__ ba1,
                      const float* __restrict__ ba2,
                      float* __restrict__ out)
{
  extern __shared__ char smem[];
  const int lane = threadIdx.x & 31;
  const int wid  = threadIdx.x >> 5;
  const int r    = lane & 15;
  const int hi   = (lane >> 4) & 1;
  const int job  = blockIdx.x * WAVES + wid;
  if (job >= NJOBS) return;          // wave-uniform: whole wave exits together
  const int eBase = job * (MT * 16);

  char* Ab  = smem + wid * LDS_A;
  char* haB = smem + WAVES * LDS_A + wid * LDS_HA;

  // ---- per-lane edge metadata: lane owns edge eBase + slot*32 + lane ----
  int srcI[2], tgtI[2], msend[2];
  float pen[2];
#pragma unroll
  for (int s = 0; s < 2; ++s) {
    int e = eBase + s * 32 + lane;
    int si = 0, ti = 0;
    if (e < NE) { si = edges[4 * e]; ti = edges[4 * e + 2]; }
    srcI[s] = si; tgtI[s] = ti;
    int sa = army[si], ta = army[ti];
    float p = 0.0f;
    if ((sa <= 2) || (ta >= 3 * sa)) p -= 1.0f;
    if (si == ti) p -= 100.0f;
    pen[s] = p;
    msend[s] = sa - 1;
  }

  // ---- gather: 64 edge rows of 512 bf16 (src||tgt embedding) into LDS ----
  // CDNA5 async global->LDS pipe: 16 B per lane per op, tracked by ASYNCcnt,
  // no VGPR data round-trip. Generic-pointer low 32 bits == LDS byte address.
  const unsigned AbOff = (unsigned)(size_t)Ab;
#pragma unroll 1
  for (int m = 0; m < MT * 16; ++m) {
    int sm = __shfl((m < 32) ? srcI[0] : srcI[1], m & 31, 32);
    int tm = __shfl((m < 32) ? tgtI[0] : tgtI[1], m & 31, 32);
    unsigned long long gs =
        (unsigned long long)(size_t)(embq + (size_t)sm * 32 + lane);
    unsigned long long gt =
        (unsigned long long)(size_t)(embq + (size_t)tm * 32 + lane);
    unsigned ls = AbOff + m * 1024 + lane * 16;
    unsigned lt = ls + 512;
#if defined(__HIP_DEVICE_COMPILE__)
    asm volatile("global_load_async_to_lds_b128 %0, %1, off"
                 :: "v"(ls), "v"(gs) : "memory");
    asm volatile("global_load_async_to_lds_b128 %0, %1, off"
                 :: "v"(lt), "v"(gt) : "memory");
#else
    ((uint4*)Ab)[m * 64 + lane]      = embq[(size_t)sm * 32 + lane];
    ((uint4*)Ab)[m * 64 + 32 + lane] = embq[(size_t)tm * 32 + lane];
#endif
  }
#if defined(__HIP_DEVICE_COMPILE__)
  asm volatile("s_wait_asynccnt 0x0" ::: "memory");  // A tile resident in LDS
#endif

  // A fragment (16x32 MxK bf16): lane r, dwords 0-7 = K kt*32+16*hi .. +15
  // ->  two b128 LDS reads.
  auto loadA = [&](int mt, int kt) -> v16bf {
    ABu u;
    const char* p = Ab + mt * 16384 + r * 1024 + (kt * 32 + hi * 16) * 2;
    u.q[0] = *(const uint4*)p;
    u.q[1] = *(const uint4*)(p + 16);
    return u.v;
  };
  auto loadB = [&](const uint4* __restrict__ wp, int tile) -> v16bf {
    ABu u;
    u.q[0] = wp[(size_t)tile * 64 + lane * 2];
    u.q[1] = wp[(size_t)tile * 64 + lane * 2 + 1];
    return u.v;
  };

  // ================= GEMM1: h = relu(E@W1 + b1); elog += h * W2 ============
  float elog[MT][8];
#pragma unroll
  for (int mt = 0; mt < MT; ++mt)
#pragma unroll
    for (int j = 0; j < 8; ++j) elog[mt][j] = 0.0f;

#pragma unroll 1
  for (int nt = 0; nt < 16; ++nt) {
    float b1v = b1[nt * 16 + r];
    float w2v = W2[nt * 16 + r];
    v8f a0 = {}, a1 = {}, a2 = {}, a3 = {};
#pragma unroll 1
    for (int kt = 0; kt < 16; ++kt) {
      v16bf bT = loadB(w1p, kt * 16 + nt);
      a0 = __builtin_amdgcn_wmma_f32_16x16x32_bf16(false, loadA(0, kt), false, bT, (short)0, a0, false, false);
      a1 = __builtin_amdgcn_wmma_f32_16x16x32_bf16(false, loadA(1, kt), false, bT, (short)0, a1, false, false);
      a2 = __builtin_amdgcn_wmma_f32_16x16x32_bf16(false, loadA(2, kt), false, bT, (short)0, a2, false, false);
      a3 = __builtin_amdgcn_wmma_f32_16x16x32_bf16(false, loadA(3, kt), false, bT, (short)0, a3, false, false);
    }
#pragma unroll
    for (int j = 0; j < 8; ++j) {
      float h;
      h = fmaxf(a0[j] + b1v, 0.0f); elog[0][j] += h * w2v;
      h = fmaxf(a1[j] + b1v, 0.0f); elog[1][j] += h * w2v;
      h = fmaxf(a2[j] + b1v, 0.0f); elog[2][j] += h * w2v;
      h = fmaxf(a3[j] + b1v, 0.0f); elog[3][j] += h * w2v;
    }
  }

  // reduce over N lanes (xor 1,2,4,8 stays inside each 16-lane half); row m = mt*16+8*hi+j
  float b2s = b2[0];
#pragma unroll
  for (int mt = 0; mt < MT; ++mt) {
#pragma unroll
    for (int j = 0; j < 8; ++j) {
      float v = elog[mt][j];
      v += __shfl_xor(v, 1, 32);
      v += __shfl_xor(v, 2, 32);
      v += __shfl_xor(v, 4, 32);
      v += __shfl_xor(v, 8, 32);
      int m = mt * 16 + hi * 8 + j;
      float p = __shfl(pen[mt >> 1], m & 31, 32);
      int e = eBase + m;
      if (r == 0 && e < NE) out[e] = v + b2s + p;
    }
  }

  // ============ GEMM2 (ha = relu(E@Wa1+ba1)) + GEMM3 (army head) ===========
#pragma unroll 1
  for (int mt = 0; mt < MT; ++mt) {
#pragma unroll 1
    for (int nt2 = 0; nt2 < 8; ++nt2) {
      v8f acc = {};
#pragma unroll 1
      for (int kt = 0; kt < 16; ++kt)
        acc = __builtin_amdgcn_wmma_f32_16x16x32_bf16(false, loadA(mt, kt), false,
                                                      loadB(wa1p, kt * 8 + nt2),
                                                      (short)0, acc, false, false);
      float bav = ba1[nt2 * 16 + r];
#pragma unroll
      for (int j = 0; j < 8; ++j) {
        unsigned hb = f2bf(fmaxf(acc[j] + bav, 0.0f));
        *(unsigned short*)(haB + (hi * 8 + j) * 256 + (nt2 * 16 + r) * 2) =
            (unsigned short)hb;
      }
    }

    int msS = (mt < 2) ? msend[0] : msend[1];
#pragma unroll 1
    for (int nt3 = 0; nt3 < 4; ++nt3) {
      v8f acc = {};
#pragma unroll
      for (int kt3 = 0; kt3 < 4; ++kt3) {
        ABu u;
        const char* p = haB + r * 256 + (kt3 * 32 + hi * 16) * 2;
        u.q[0] = *(const uint4*)p;
        u.q[1] = *(const uint4*)(p + 16);
        acc = __builtin_amdgcn_wmma_f32_16x16x32_bf16(false, u.v, false,
                                                      loadB(wa2p, kt3 * 4 + nt3),
                                                      (short)0, acc, false, false);
      }
      int col = nt3 * 16 + r;
      float bav = ba2[col];
#pragma unroll
      for (int j = 0; j < 8; ++j) {
        int m = mt * 16 + hi * 8 + j;
        int e = eBase + m;
        int msd = __shfl(msS, m & 31, 32);
        float v = acc[j] + bav;
        v = (col <= msd) ? v : -1.0e9f;
        if (e < NE) out[(size_t)NE + (size_t)e * 64 + col] = v;
      }
    }
  }
}

extern "C" void kernel_launch(void* const* d_in, const int* in_sizes, int n_in,
                              void* d_out, int out_size, void* d_ws, size_t ws_size,
                              hipStream_t stream) {
  const float* nodeEmb = (const float*)d_in[0];
  const int*   edges   = (const int*)d_in[1];   // int64 [E][2] viewed as int32
  const int*   army    = (const int*)d_in[2];
  const float* W1  = (const float*)d_in[3];
  const float* b1  = (const float*)d_in[4];
  const float* W2  = (const float*)d_in[5];
  const float* b2  = (const float*)d_in[6];
  const float* Wa1 = (const float*)d_in[7];
  const float* ba1 = (const float*)d_in[8];
  const float* Wa2 = (const float*)d_in[9];
  const float* ba2 = (const float*)d_in[10];
  float* out = (float*)d_out;

  char* ws = (char*)d_ws;
  const size_t embBytes = (size_t)100000 * 256 * 2;          // 51.2 MB bf16 table
  uint4*    embq = (uint4*)ws;
  unsigned* w1p  = (unsigned*)(ws + embBytes);               // 256 KB packed W1
  unsigned* wa1p = w1p + 16 * 16 * 256;                      // 128 KB packed Wa1
  unsigned* wa2p = wa1p + 16 * 8 * 256;                      // 16 KB packed Wa2

  const int n4 = 100000 * 256 / 4;
  conv_emb_kernel<<<(n4 + 255) / 256, 256, 0, stream>>>(
      (const float4*)nodeEmb, (uint2*)ws, n4);
  pack_w_kernel<<<(16 * 16 * 256 + 255) / 256, 256, 0, stream>>>(W1, 512, 256, w1p);
  pack_w_kernel<<<(16 * 8 * 256 + 255) / 256, 256, 0, stream>>>(Wa1, 512, 128, wa1p);
  pack_w_kernel<<<(4 * 4 * 256 + 255) / 256, 256, 0, stream>>>(Wa2, 128, 64, wa2p);

  attack_head_main<<<GRID_MAIN, WAVES * 32, LDS_TOTAL, stream>>>(
      embq, edges, army,
      (const uint4*)w1p, (const uint4*)wa1p, (const uint4*)wa2p,
      b1, W2, b2, ba1, ba2, out);
}
